// AttentionLayer1_71863392796936
// MI455X (gfx1250) — matrix operand order repaired and checked by reading, not compiled, not run on hardware
//
#include <hip/hip_runtime.h>
#include <hip/hip_bf16.h>

typedef __bf16 bf16_t;
typedef __attribute__((ext_vector_type(16))) __bf16 v16bf;
typedef __attribute__((ext_vector_type(8)))  float  v8f;

#define DM    512
#define BB    32
#define LL    512
#define NHD   8
#define HDM   64
#define FOURH 2048

// ---------------------------------------------------------------------------
// WMMA fragment helpers (wave32, CDNA5 16x16x32 bf16; A and B both use contiguous K = 16*(lane/16) + i)
// ---------------------------------------------------------------------------

__device__ __forceinline__ v8f v8f_zero() {
  v8f z;
#pragma unroll
  for (int i = 0; i < 8; ++i) z[i] = 0.0f;
  return z;
}

// A fragment 16x32 (MxK) from row-major storage, leading dim ldr (elements).
// lane<16: M=lane, K = {0..15}; lane>=16: M=lane-16, K = {16..31}
__device__ __forceinline__ v16bf load_a_rowmajor(const bf16_t* p, int ldr) {
  const int lane = threadIdx.x & 31;
  const int row  = lane & 15;
  const int ko   = (lane >> 4) * 16;
  const bf16_t* r = p + (size_t)row * ldr + ko;
  v16bf a;
#pragma unroll
  for (int i = 0; i < 8; ++i) a[i] = r[i];
#pragma unroll
  for (int i = 0; i < 8; ++i) a[8 + i] = r[8 + i];
  return a;
}

// B fragment 32x16 (KxN) where B = W^T and W is row-major [N][K] (ldw = K
// pitch). Per lane this is 16 *contiguous* bf16 from one row of W.
__device__ __forceinline__ v16bf load_b_wT(const bf16_t* w, int n0, int k0, int ldw) {
  const int lane = threadIdx.x & 31;
  const int n  = n0 + (lane & 15);
  const int kb = k0 + (lane >> 4) * 16;
  const bf16_t* r = w + (size_t)n * ldw + kb;
  v16bf b;
#pragma unroll
  for (int i = 0; i < 16; ++i) b[i] = r[i];
  return b;
}

// B fragment 32x16 (KxN) from natural row-major [K][N] storage (strided).
__device__ __forceinline__ v16bf load_b_rowmajor(const bf16_t* p, int k0, int n0, int ldb) {
  const int lane = threadIdx.x & 31;
  const int n  = n0 + (lane & 15);
  const int kb = k0 + (lane >> 4) * 16;
  const bf16_t* r = p + (size_t)kb * ldb + n;
  v16bf b;
#pragma unroll
  for (int i = 0; i < 16; ++i) b[i] = r[(size_t)i * ldb];
  return b;
}

__device__ __forceinline__ v8f wmma_bf16(v16bf a, v16bf b, v8f c) {
  return __builtin_amdgcn_wmma_f32_16x16x32_bf16(false, a, false, b, (short)0, c,
                                                 false, false);
}

__device__ __forceinline__ float fsigmoid(float x) {
  return 1.0f / (1.0f + __expf(-x));
}
__device__ __forceinline__ float ftanh(float x) {
  return 1.0f - 2.0f / (__expf(2.0f * x) + 1.0f);
}

// ---------------------------------------------------------------------------
// Conversion kernels
// ---------------------------------------------------------------------------
__global__ void cvt_f32_bf16_kernel(const float* __restrict__ src,
                                    bf16_t* __restrict__ dst, int n) {
  int i = blockIdx.x * blockDim.x + threadIdx.x;
  int stride = gridDim.x * blockDim.x;
  for (; i < n; i += stride) dst[i] = (bf16_t)src[i];
}

__global__ void bias_sum_kernel(const float* __restrict__ a,
                                const float* __restrict__ b,
                                float* __restrict__ dst, int n) {
  int i = blockIdx.x * blockDim.x + threadIdx.x;
  if (i < n) dst[i] = a[i] + b[i];
}

// ---------------------------------------------------------------------------
// LSTM recurrence: 6 blocks = 3 streams x 2 batch-halves (M=16). 16 waves per
// block; wave w owns hidden columns [32w, 32w+32) across all 4 gates.
// gx folded into each step (same lanes, no GX buffer). c stays in f32 regs.
// ---------------------------------------------------------------------------
#define HROW (DM + 8)   // LDS row pitch: 4-bank rotation per row -> no conflicts

__global__ __launch_bounds__(512)
void lstm_recurrence_kernel(const bf16_t* __restrict__ xb,   // [3][B][L][D] bf16
                            const bf16_t* __restrict__ wih,  // [3][4H][D]
                            const bf16_t* __restrict__ whh,  // [3][4H][D]
                            const float*  __restrict__ bias, // [3][4H]
                            bf16_t* __restrict__ hs)         // [3][B][L][D] bf16
{
  const int s     = blockIdx.x >> 1;
  const int b0    = (blockIdx.x & 1) * 16;
  const int wave  = threadIdx.x >> 5;
  const int lane  = threadIdx.x & 31;
  const int ln16  = lane & 15;
  const int rbase = (lane >> 4) * 8;

  const size_t BLD = (size_t)BB * LL * DM;
  const bf16_t* xs  = xb  + (size_t)s * BLD + (size_t)b0 * LL * DM;
  const bf16_t* wis = wih + (size_t)s * FOURH * DM;
  const bf16_t* whs = whh + (size_t)s * FOURH * DM;
  const float*  bs  = bias + s * FOURH;
  bf16_t* hss = hs + (size_t)s * BLD + (size_t)b0 * LL * DM;

  __shared__ bf16_t Hs[16 * HROW];
  for (int i = threadIdx.x; i < 16 * HROW; i += blockDim.x) Hs[i] = (bf16_t)0.0f;

  float c_reg[2][8];
#pragma unroll
  for (int ct = 0; ct < 2; ++ct)
#pragma unroll
    for (int r = 0; r < 8; ++r) c_reg[ct][r] = 0.0f;

  const int hc0 = wave * 32;

  for (int t = 0; t < LL; ++t) {
    v8f acc[4][2];
#pragma unroll
    for (int g = 0; g < 4; ++g)
#pragma unroll
      for (int ct = 0; ct < 2; ++ct) acc[g][ct] = v8f_zero();

    // ---- input contribution x_t @ W_ih^T (independent of Hs: before barrier)
    const bf16_t* xrow = xs + (size_t)t * DM;
    if (t + 1 < LL)  // pull next timestep's activations toward the caches
      __builtin_prefetch(xs + (size_t)ln16 * LL * DM + (size_t)(t + 1) * DM, 0, 1);
#pragma unroll 1
    for (int k0 = 0; k0 < DM; k0 += 32) {
      v16bf a = load_a_rowmajor(xrow + k0, LL * DM);
#pragma unroll
      for (int g = 0; g < 4; ++g)
#pragma unroll
        for (int ct = 0; ct < 2; ++ct) {
          v16bf bb = load_b_wT(wis, g * DM + hc0 + ct * 16, k0, DM);
          acc[g][ct] = wmma_bf16(a, bb, acc[g][ct]);
        }
    }

    __syncthreads();  // Hs writes from step t-1 are visible

    // ---- hidden contribution h_{t-1} @ W_hh^T
#pragma unroll 1
    for (int k0 = 0; k0 < DM; k0 += 32) {
      v16bf a = load_a_rowmajor(Hs + k0, HROW);
#pragma unroll
      for (int g = 0; g < 4; ++g)
#pragma unroll
        for (int ct = 0; ct < 2; ++ct) {
          v16bf bb = load_b_wT(whs, g * DM + hc0 + ct * 16, k0, DM);
          acc[g][ct] = wmma_bf16(a, bb, acc[g][ct]);
        }
    }

    __syncthreads();  // all waves done reading Hs

    // ---- gates: i/f/g/o tiles share (row, col) per lane in D-layout
#pragma unroll
    for (int ct = 0; ct < 2; ++ct) {
      const int n = hc0 + ct * 16 + ln16;
#pragma unroll
      for (int r = 0; r < 8; ++r) {
        const int row = rbase + r;
        float iv = acc[0][ct][r] + bs[0 * DM + n];
        float fv = acc[1][ct][r] + bs[1 * DM + n];
        float gv = acc[2][ct][r] + bs[2 * DM + n];
        float ov = acc[3][ct][r] + bs[3 * DM + n];
        float c  = fsigmoid(fv) * c_reg[ct][r] + fsigmoid(iv) * ftanh(gv);
        float hv = fsigmoid(ov) * ftanh(c);
        c_reg[ct][r] = c;
        bf16_t hb = (bf16_t)hv;
        Hs[row * HROW + n] = hb;
        hss[(size_t)row * LL * DM + (size_t)t * DM + n] = hb;
      }
    }
  }
}

// ---------------------------------------------------------------------------
// Flash attention: one wave per (batch, head, q-tile of 16). Online softmax
// over 512 keys in chunks of 32 (two 16x16 S tiles -> one K=32 P.V WMMA).
// ---------------------------------------------------------------------------
#define PROW 40  // LDS pitch for P staging (conflict-avoiding)

__global__ __launch_bounds__(256)
void attention_kernel(const bf16_t* __restrict__ hs,  // [3][B][L][D] bf16
                      bf16_t* __restrict__ ob)        // [B][L][D]   bf16
{
  const int wave  = threadIdx.x >> 5;
  const int lane  = threadIdx.x & 31;
  const int ln16  = lane & 15;
  const int rbase = (lane >> 4) * 8;
  const int gw = blockIdx.x * 8 + wave;
  const int qt = gw & 31;          // q tile (L/16 = 32)
  const int h  = (gw >> 5) & 7;    // head
  const int b  = gw >> 8;          // batch

  const size_t BLD = (size_t)BB * LL * DM;
  const bf16_t* hq = hs + (size_t)b * LL * DM;
  const bf16_t* hk = hs + BLD + (size_t)b * LL * DM;
  const bf16_t* hv = hs + 2 * BLD + (size_t)b * LL * DM;

  __shared__ bf16_t Ps[8 * 16 * PROW];
  bf16_t* ps = Ps + wave * 16 * PROW;

  // Q fragments for K(head-dim) = 64: two A fragments
  v16bf aq0 = load_a_rowmajor(hq + (size_t)qt * 16 * DM + h * HDM + 0,  DM);
  v16bf aq1 = load_a_rowmajor(hq + (size_t)qt * 16 * DM + h * HDM + 32, DM);

  v8f o_acc[4];
#pragma unroll
  for (int nt = 0; nt < 4; ++nt) o_acc[nt] = v8f_zero();
  float m_i[8], l_i[8];
#pragma unroll
  for (int r = 0; r < 8; ++r) { m_i[r] = -1e30f; l_i[r] = 0.0f; }

  for (int kt2 = 0; kt2 < LL / 32; ++kt2) {
    v8f st[2];
#pragma unroll
    for (int hh = 0; hh < 2; ++hh) {
      const int kt = kt2 * 2 + hh;
      v8f sacc = v8f_zero();
      v16bf bk0 = load_b_wT(hk + h * HDM, kt * 16, 0,  DM);  // contiguous reads
      sacc = wmma_bf16(aq0, bk0, sacc);
      v16bf bk1 = load_b_wT(hk + h * HDM, kt * 16, 32, DM);
      sacc = wmma_bf16(aq1, bk1, sacc);
      st[hh] = sacc;
    }
    // online softmax; rows of D-layout live in half-wave lane groups (width 16)
#pragma unroll
    for (int r = 0; r < 8; ++r) {
      float s0 = st[0][r] * 0.125f;   // 1/sqrt(64)
      float s1 = st[1][r] * 0.125f;
      float tm = fmaxf(s0, s1);
#pragma unroll
      for (int msk = 8; msk >= 1; msk >>= 1)
        tm = fmaxf(tm, __shfl_xor(tm, msk, 16));
      float mnew = fmaxf(m_i[r], tm);
      float sc = __expf(m_i[r] - mnew);
      float p0 = __expf(s0 - mnew);
      float p1 = __expf(s1 - mnew);
      float rs = p0 + p1;
#pragma unroll
      for (int msk = 8; msk >= 1; msk >>= 1)
        rs += __shfl_xor(rs, msk, 16);
      l_i[r] = l_i[r] * sc + rs;
      m_i[r] = mnew;
#pragma unroll
      for (int nt = 0; nt < 4; ++nt) o_acc[nt][r] *= sc;
      const int row = rbase + r;
      ps[row * PROW + 0  + ln16] = (bf16_t)p0;   // D-layout -> A-layout via LDS
      ps[row * PROW + 16 + ln16] = (bf16_t)p1;   // (same-wave DS ops in-order)
    }
    v16bf ap = load_a_rowmajor(ps, PROW);
#pragma unroll
    for (int nt = 0; nt < 4; ++nt) {
      v16bf bv = load_b_rowmajor(hv + h * HDM, kt2 * 32, nt * 16, DM);
      o_acc[nt] = wmma_bf16(ap, bv, o_acc[nt]);
    }
  }

  bf16_t* ob_b = ob + (size_t)b * LL * DM;
#pragma unroll
  for (int nt = 0; nt < 4; ++nt)
#pragma unroll
    for (int r = 0; r < 8; ++r) {
      int row = qt * 16 + rbase + r;
      int col = h * HDM + nt * 16 + ln16;
      ob_b[(size_t)row * DM + col] = (bf16_t)(o_acc[nt][r] / l_i[r]);
    }
}

// ---------------------------------------------------------------------------
// Output projection: OUT = O @ W_out^T + b_out (f32 out)
// ---------------------------------------------------------------------------
__global__ __launch_bounds__(256)
void proj_kernel(const bf16_t* __restrict__ ob,   // [B*L][D] bf16
                 const bf16_t* __restrict__ wo,   // [D][D]  bf16
                 const float*  __restrict__ bo,   // [D]
                 float* __restrict__ out)         // [B*L][D]
{
  const int wave  = threadIdx.x >> 5;
  const int lane  = threadIdx.x & 31;
  const int ln16  = lane & 15;
  const int rbase = (lane >> 4) * 8;
  const int m0 = blockIdx.x * 16;
  const int n0 = blockIdx.y * 128 + wave * 16;

  v8f acc = v8f_zero();
#pragma unroll 1
  for (int k0 = 0; k0 < DM; k0 += 32) {
    v16bf a  = load_a_rowmajor(ob + (size_t)m0 * DM + k0, DM);
    v16bf bb = load_b_wT(wo, n0, k0, DM);
    acc = wmma_bf16(a, bb, acc);
  }
#pragma unroll
  for (int r = 0; r < 8; ++r) {
    int row = m0 + rbase + r;
    int col = n0 + ln16;
    out[(size_t)row * DM + col] = acc[r] + bo[col];
  }
}

// ---------------------------------------------------------------------------
// Host launcher
// ---------------------------------------------------------------------------
extern "C" void kernel_launch(void* const* d_in, const int* in_sizes, int n_in,
                              void* d_out, int out_size, void* d_ws, size_t ws_size,
                              hipStream_t stream) {
  (void)in_sizes; (void)n_in; (void)out_size; (void)ws_size;

  const float* q    = (const float*)d_in[0];
  const float* k    = (const float*)d_in[1];
  const float* v    = (const float*)d_in[2];
  const float* wih[3] = {(const float*)d_in[3], (const float*)d_in[7],  (const float*)d_in[11]};
  const float* whh[3] = {(const float*)d_in[4], (const float*)d_in[8],  (const float*)d_in[12]};
  const float* bih[3] = {(const float*)d_in[5], (const float*)d_in[9],  (const float*)d_in[13]};
  const float* bhh[3] = {(const float*)d_in[6], (const float*)d_in[10], (const float*)d_in[14]};
  const float* w_out = (const float*)d_in[15];
  const float* b_out = (const float*)d_in[16];

  const size_t BLD = (size_t)BB * LL * DM;        // 8,388,608 elems
  const size_t WSZ = (size_t)FOURH * DM;          // 1,048,576 elems

  // workspace layout (bf16 unless noted), ~125 MiB total
  char* w = (char*)d_ws;
  bf16_t* xb   = (bf16_t*)w;                 w += 3 * BLD * sizeof(bf16_t);
  bf16_t* hsb  = (bf16_t*)w;                 w += 3 * BLD * sizeof(bf16_t);
  bf16_t* wihb = (bf16_t*)w;                 w += 3 * WSZ * sizeof(bf16_t);
  bf16_t* whhb = (bf16_t*)w;                 w += 3 * WSZ * sizeof(bf16_t);
  bf16_t* wob  = (bf16_t*)w;                 w += (size_t)DM * DM * sizeof(bf16_t);
  float*  biasb = (float*)w;                 w += 3 * FOURH * sizeof(float);
  bf16_t* obuf = (bf16_t*)w;

  const float* xin[3] = {q, k, v};
  for (int s = 0; s < 3; ++s) {
    cvt_f32_bf16_kernel<<<4096, 256, 0, stream>>>(xin[s], xb + s * BLD, (int)BLD);
    cvt_f32_bf16_kernel<<<2048, 256, 0, stream>>>(wih[s], wihb + s * WSZ, (int)WSZ);
    cvt_f32_bf16_kernel<<<2048, 256, 0, stream>>>(whh[s], whhb + s * WSZ, (int)WSZ);
    bias_sum_kernel<<<(FOURH + 255) / 256, 256, 0, stream>>>(bih[s], bhh[s],
                                                             biasb + s * FOURH, FOURH);
  }
  cvt_f32_bf16_kernel<<<1024, 256, 0, stream>>>(w_out, wob, DM * DM);

  lstm_recurrence_kernel<<<6, 512, 0, stream>>>(xb, wihb, whhb, biasb, hsb);

  attention_kernel<<<1024, 256, 0, stream>>>(hsb, obuf);

  proj_kernel<<<dim3(1024, 4), 256, 0, stream>>>(obuf, wob, b_out, (float*)d_out);
}
